// TransformerDecoder_8383776162585
// MI455X (gfx1250) — matrix operand order repaired and checked by reading, not compiled, not run on hardware
//
#include <hip/hip_runtime.h>
#include <hip/hip_bf16.h>
#include <math.h>

// ---------------------------------------------------------------------------
// CDNA5 / gfx1250 transformer decoder forward.
// All GEMMs + attention run on v_wmma_f32_16x16x32_f16 (f16 in, f32 acc).
// Attention is flash-style (no [B,H,T,T] HBM round trip).
// Register-staged double buffering: next tile's global_load_b128s are issued
// before the WMMA phase; conversion/LDS store waits only at the next tile.
// f32->f16 staging uses v_cvt_pk_rtz_f16_f32 (packed converts).
// ---------------------------------------------------------------------------

typedef _Float16 v16h __attribute__((ext_vector_type(16)));
typedef __fp16   v2fp __attribute__((ext_vector_type(2)));   // pkrtz ret type
typedef float    v8f  __attribute__((ext_vector_type(8)));

#define D_MODEL 1024
#define NHEAD   16
#define HDIM    64
#define SEQ     2048
#define NBATCH  2
#define NLAYER  6
#define FFDIM   2048
#define NVOCAB  99
#define MROWS   (NBATCH * SEQ)   // 4096 token rows

union H4 { v2fp p[2]; unsigned long long u; };
__device__ __forceinline__ unsigned long long pack4(float4 f) {
  H4 cv;
  cv.p[0] = __builtin_amdgcn_cvt_pkrtz(f.x, f.y);   // v_cvt_pk_rtz_f16_f32
  cv.p[1] = __builtin_amdgcn_cvt_pkrtz(f.z, f.w);
  return cv.u;
}

// ---------------------------------------------------------------------------
// WMMA fragment helpers (ISA 7.12.2 VGPR layouts, wave32).
// A 16x32 f16: lane holds row m = lane&15; element e (vgpr j=e>>1, half h=e&1)
//   K = 16*(j>=4) + 8*(lane>=16) + 2*(j&3) + h
// B 32x16 f16: lane holds col n = lane&15; element e: K = 16*(e>=8) + 8*(lane>=16) + (e&7)
// C/D 16x16 f32: lane holds col n = lane&15; vgpr r: row m = r + 8*(lane>=16)
// ---------------------------------------------------------------------------
__device__ __forceinline__ int ka_map(int e, int lane) {
  int j = e >> 1, h = e & 1;
  return ((j & 4) << 2) | ((lane & 16) >> 1) | ((j & 3) << 1) | h;
}
__device__ __forceinline__ int kb_map(int e, int lane) {
  return ((e & 8) << 1) | ((lane & 16) >> 1) | (e & 7);
}
__device__ __forceinline__ v8f wmma_f16(v16h a, v16h b, v8f c) {
  return __builtin_amdgcn_wmma_f32_16x16x32_f16(
      false, a, false, b, (short)0, c, false, false);
}
__device__ __forceinline__ v16h frag_a(const _Float16* s, int row0, int stride,
                                       int lane, int kb) {
  v16h f;
  int m = row0 + (lane & 15);
#pragma unroll
  for (int e = 0; e < 16; ++e) f[e] = s[m * stride + kb + ka_map(e, lane)];
  return f;
}
__device__ __forceinline__ v16h frag_b_nk(const _Float16* s, int n0, int stride,
                                          int lane, int kb) {
  v16h f;
  int n = n0 + (lane & 15);
#pragma unroll
  for (int e = 0; e < 16; ++e) f[e] = s[n * stride + kb + kb_map(e, lane)];
  return f;
}
__device__ __forceinline__ v16h frag_b_kn(const _Float16* s, int n0, int stride,
                                          int lane, int kb) {
  v16h f;
  int n = n0 + (lane & 15);
#pragma unroll
  for (int e = 0; e < 16; ++e) f[e] = s[(kb + kb_map(e, lane)) * stride + n];
  return f;
}

// ---------------------------------------------------------------------------
// Positional encoding:  x[b,t,:] = tokens[b,t,:] + PE(t,:)
// ---------------------------------------------------------------------------
__global__ __launch_bounds__(256) void posenc_kernel(
    const float* __restrict__ tok, float* __restrict__ x) {
  int row = blockIdx.x;            // b*T + t
  int t = row & (SEQ - 1);
#pragma unroll
  for (int it = 0; it < D_MODEL / 256; ++it) {
    int i = threadIdx.x + it * 256;
    float dv = __powf(10000.0f, 2.0f * (float)i / (float)D_MODEL);
    float ang = (float)t / dv;
    float pe = (i & 1) ? __cosf(ang) : __sinf(ang);
    x[(size_t)row * D_MODEL + i] = tok[(size_t)row * D_MODEL + i] + pe;
  }
}

// ---------------------------------------------------------------------------
// GEMM:  C[M,N] = A[M,K] @ W[N,K]^T + bias (+ optional ReLU)
// Block tile 128x128, 8 waves, each wave a 32x64 patch = 2x4 WMMA tiles.
// Register-staged pipeline: loads for tile k+1 issued before tile k's WMMAs.
// ---------------------------------------------------------------------------
#define BMG 128
#define BNG 128
#define AST 36   // padded LDS stride (halves): 18*m mod 64 distinct, 8B aligned
__global__ __launch_bounds__(256) void gemm_f16w(
    const float* __restrict__ A, const float* __restrict__ W,
    const float* __restrict__ bias, float* __restrict__ C,
    int M, int N, int K, int relu) {
  __shared__ _Float16 As[BMG * AST];
  __shared__ _Float16 Bs[BNG * AST];
  int tid = threadIdx.x, lane = tid & 31, wid = tid >> 5;
  int wr = wid & 3, wc = wid >> 2;               // 4 M-groups x 2 N-groups
  int m0 = blockIdx.y * BMG, n0 = blockIdx.x * BNG;
  int arow = tid >> 3, aquad = tid & 7;          // 32 rows/pass, 8 quads/row
  bool fullN = (n0 + BNG) <= N;                  // block-uniform guard
  size_t aBase = (size_t)(m0 + arow) * (size_t)K + (size_t)(aquad * 4);
  size_t wBase = (size_t)(n0 + arow) * (size_t)K + (size_t)(aquad * 4);
  size_t rowStep = (size_t)32 * (size_t)K;

  float4 ra[4], rw[4];
  v8f acc[2][4] = {};

  auto loadTile = [&](int k0) {
#pragma unroll
    for (int it = 0; it < 4; ++it)
      ra[it] = *reinterpret_cast<const float4*>(&A[aBase + it * rowStep + k0]);
    if (fullN) {
#pragma unroll
      for (int it = 0; it < 4; ++it)
        rw[it] =
            *reinterpret_cast<const float4*>(&W[wBase + it * rowStep + k0]);
    } else {
#pragma unroll
      for (int it = 0; it < 4; ++it) {
        if (n0 + arow + it * 32 < N)
          rw[it] =
              *reinterpret_cast<const float4*>(&W[wBase + it * rowStep + k0]);
        else
          rw[it] = make_float4(0.f, 0.f, 0.f, 0.f);
      }
    }
  };
  auto stageTile = [&]() {
#pragma unroll
    for (int it = 0; it < 4; ++it) {
      int r = arow + it * 32;
      *reinterpret_cast<unsigned long long*>(&As[r * AST + aquad * 4]) =
          pack4(ra[it]);
      *reinterpret_cast<unsigned long long*>(&Bs[r * AST + aquad * 4]) =
          pack4(rw[it]);
    }
  };

  loadTile(0);
  for (int k0 = 0; k0 < K; k0 += 32) {
    stageTile();                       // waits on loads issued last iteration
    __syncthreads();
    if (k0 + 32 < K) loadTile(k0 + 32);   // in flight during WMMA phase
    if (k0 + 64 < K)                      // L2 warm-up two tiles ahead
      __builtin_prefetch(&A[aBase + k0 + 64], 0, 3);

    v16h a0 = frag_a(As, 32 * wr, AST, lane, 0);
    v16h a1 = frag_a(As, 32 * wr + 16, AST, lane, 0);
#pragma unroll
    for (int j = 0; j < 4; ++j) {
      v16h bj = frag_b_nk(Bs, 64 * wc + 16 * j, AST, lane, 0);
      acc[0][j] = wmma_f16(a0, bj, acc[0][j]);
      acc[1][j] = wmma_f16(a1, bj, acc[1][j]);
    }
    __syncthreads();
  }

  int crow = (lane & 16) >> 1, ccol = lane & 15;
#pragma unroll
  for (int j = 0; j < 4; ++j) {
    int n = n0 + 64 * wc + 16 * j + ccol;
    if (fullN || n < N) {
      float bn = bias[n];
#pragma unroll
      for (int i = 0; i < 2; ++i) {
#pragma unroll
        for (int r = 0; r < 8; ++r) {
          int m = m0 + 32 * wr + 16 * i + r + crow;
          float val = acc[i][j][r] + bn;
          if (relu) val = fmaxf(val, 0.0f);
          C[(size_t)m * N + n] = val;
        }
      }
    }
  }
}

// ---------------------------------------------------------------------------
// Fused causal multi-head attention (flash-style).
// Grid: (T/64, B*H). Block: 128 threads = 4 waves; wave w owns q rows
// [q0+16w, q0+16w+16). Streams 64-key tiles: S=Q@K^T (WMMA) -> LDS ->
// online softmax (2 threads/row) -> acc = acc*rescale + P@V (WMMA).
// K/V tiles are register-staged one iteration ahead.
// ---------------------------------------------------------------------------
#define QST 68   // padded half stride, 8B-aligned rows, conflict-free frags
#define SST 65   // padded float stride for score tile
#define PST 66   // padded half stride for P tile
__global__ __launch_bounds__(128) void attention_fused(
    const float* __restrict__ Q, const float* __restrict__ Kx,
    const float* __restrict__ Vx, float* __restrict__ O) {
  __shared__ _Float16 Qs[64 * QST];
  __shared__ _Float16 Ks[64 * QST];
  __shared__ _Float16 Vs[64 * QST];
  __shared__ float    Sb[64 * SST];
  __shared__ _Float16 Ps[64 * PST];
  __shared__ float Mrow[64], Lrow[64], Scl[64];
  __shared__ float PM[128], PSu[128];

  int tid = threadIdx.x, lane = tid & 31, wid = tid >> 5;
  int q0 = blockIdx.x * 64;
  int bh = blockIdx.y;
  int b = bh >> 4, h = bh & 15;          // H = 16
  const float scale = 0.125f;            // 1/sqrt(HDIM=64)
  int crow = (lane & 16) >> 1, ccol = lane & 15;
  int lrow = tid >> 4, lquad = tid & 15; // staging coords: 8 rows/pass
  size_t kvBase =
      (size_t)(b * SEQ + lrow) * D_MODEL + (size_t)(h * HDIM + lquad * 4);

#pragma unroll
  for (int it = 0; it < 8; ++it) {
    int r = lrow + it * 8;
    float4 f = *reinterpret_cast<const float4*>(
        &Q[(size_t)(b * SEQ + q0 + r) * D_MODEL + h * HDIM + lquad * 4]);
    *reinterpret_cast<unsigned long long*>(&Qs[r * QST + lquad * 4]) = pack4(f);
  }
  if (tid < 64) { Mrow[tid] = -1e30f; Lrow[tid] = 0.0f; Scl[tid] = 1.0f; }

  float4 rk[8], rv[8];
  auto loadKV = [&](int kt) {
#pragma unroll
    for (int it = 0; it < 8; ++it) {
      size_t g = kvBase + (size_t)(kt + it * 8) * D_MODEL;
      rk[it] = *reinterpret_cast<const float4*>(&Kx[g]);
      rv[it] = *reinterpret_cast<const float4*>(&Vx[g]);
    }
  };
  auto stageKV = [&]() {
#pragma unroll
    for (int it = 0; it < 8; ++it) {
      int r = lrow + it * 8;
      *reinterpret_cast<unsigned long long*>(&Ks[r * QST + lquad * 4]) =
          pack4(rk[it]);
      *reinterpret_cast<unsigned long long*>(&Vs[r * QST + lquad * 4]) =
          pack4(rv[it]);
    }
  };

  v8f acc0 = {}, acc1 = {}, acc2 = {}, acc3 = {};
  loadKV(0);

  for (int kt = 0; kt <= q0; kt += 64) {   // causal: only key tiles <= diag
    stageKV();                 // prior iteration's trailing barrier protects
    __syncthreads();
    if (kt + 64 <= q0) loadKV(kt + 64);    // in flight during compute phases

    // S(16q x 64keys) = Q @ K^T ;  B[k=j][n=key] = K[key][j] -> [n][k] layout
    v8f s0 = {}, s1 = {}, s2 = {}, s3 = {};
#pragma unroll
    for (int kb = 0; kb < 64; kb += 32) {
      v16h aq = frag_a(Qs, 16 * wid, QST, lane, kb);
      s0 = wmma_f16(aq, frag_b_nk(Ks, 0, QST, lane, kb), s0);
      s1 = wmma_f16(aq, frag_b_nk(Ks, 16, QST, lane, kb), s1);
      s2 = wmma_f16(aq, frag_b_nk(Ks, 32, QST, lane, kb), s2);
      s3 = wmma_f16(aq, frag_b_nk(Ks, 48, QST, lane, kb), s3);
    }
#pragma unroll
    for (int nt = 0; nt < 4; ++nt) {
      v8f sv = (nt == 0) ? s0 : (nt == 1) ? s1 : (nt == 2) ? s2 : s3;
      int key = nt * 16 + ccol;
#pragma unroll
      for (int r = 0; r < 8; ++r) {
        int m = 16 * wid + r + crow;
        float val = sv[r] * scale;
        if (kt + key > q0 + m) val = -1e30f;   // causal mask (finite sentinel)
        Sb[m * SST + key] = val;
      }
    }
    __syncthreads();

    // online softmax: 2 threads per query row (32 keys each)
    {
      int rr = tid >> 1, hf = tid & 1;
      float tmax = -1e30f;
#pragma unroll 8
      for (int kk = 0; kk < 32; ++kk)
        tmax = fmaxf(tmax, Sb[rr * SST + hf * 32 + kk]);
      PM[tid] = tmax;
      __syncthreads();
      float mnew = fmaxf(Mrow[rr], fmaxf(PM[rr * 2], PM[rr * 2 + 1]));
      float psum = 0.f;
#pragma unroll 8
      for (int kk = 0; kk < 32; ++kk) {
        float p = __expf(Sb[rr * SST + hf * 32 + kk] - mnew);
        Ps[rr * PST + hf * 32 + kk] = (_Float16)p;
        psum += p;
      }
      PSu[tid] = psum;
      __syncthreads();
      if (tid < 64) {
        float mold = Mrow[tid];
        float mn = fmaxf(mold, fmaxf(PM[tid * 2], PM[tid * 2 + 1]));
        float sc = __expf(mold - mn);
        Lrow[tid] = Lrow[tid] * sc + PSu[tid * 2] + PSu[tid * 2 + 1];
        Mrow[tid] = mn;
        Scl[tid] = sc;
      }
      __syncthreads();
    }

    // rescale accumulators, then acc += P @ V  (V tile is [k=key][n=j])
    {
      float sl[8];
#pragma unroll
      for (int r = 0; r < 8; ++r) sl[r] = Scl[16 * wid + r + crow];
#pragma unroll
      for (int r = 0; r < 8; ++r) {
        acc0[r] *= sl[r]; acc1[r] *= sl[r]; acc2[r] *= sl[r]; acc3[r] *= sl[r];
      }
    }
#pragma unroll
    for (int kb = 0; kb < 64; kb += 32) {
      v16h ap = frag_a(Ps, 16 * wid, PST, lane, kb);
      acc0 = wmma_f16(ap, frag_b_kn(Vs, 0, QST, lane, kb), acc0);
      acc1 = wmma_f16(ap, frag_b_kn(Vs, 16, QST, lane, kb), acc1);
      acc2 = wmma_f16(ap, frag_b_kn(Vs, 32, QST, lane, kb), acc2);
      acc3 = wmma_f16(ap, frag_b_kn(Vs, 48, QST, lane, kb), acc3);
    }
    __syncthreads();
  }

  // O = acc / l
#pragma unroll
  for (int r = 0; r < 8; ++r) {
    int m = 16 * wid + r + crow;
    float inv = 1.0f / Lrow[m];
    size_t base = (size_t)(b * SEQ + q0 + m) * D_MODEL + h * HDIM;
    O[base + 0 + ccol]  = acc0[r] * inv;
    O[base + 16 + ccol] = acc1[r] * inv;
    O[base + 32 + ccol] = acc2[r] * inv;
    O[base + 48 + ccol] = acc3[r] * inv;
  }
}

// ---------------------------------------------------------------------------
// x = LayerNorm(x + y) * g + b     (one block per token row, D = 1024)
// ---------------------------------------------------------------------------
__global__ __launch_bounds__(256) void ln_residual(
    float* __restrict__ x, const float* __restrict__ y,
    const float* __restrict__ g, const float* __restrict__ bta) {
  __shared__ float red[256];
  __shared__ float stat;
  int row = blockIdx.x, tid = threadIdx.x;
  float4 vx =
      *reinterpret_cast<const float4*>(&x[(size_t)row * D_MODEL + tid * 4]);
  float4 vy =
      *reinterpret_cast<const float4*>(&y[(size_t)row * D_MODEL + tid * 4]);
  float v[4] = {vx.x + vy.x, vx.y + vy.y, vx.z + vy.z, vx.w + vy.w};
  float s = v[0] + v[1] + v[2] + v[3];
  red[tid] = s; __syncthreads();
  for (int o = 128; o > 0; o >>= 1) {
    if (tid < o) red[tid] += red[tid + o];
    __syncthreads();
  }
  if (tid == 0) stat = red[0] * (1.0f / D_MODEL);
  __syncthreads();
  float mu = stat;
  s = 0.f;
#pragma unroll
  for (int i = 0; i < 4; ++i) { float d = v[i] - mu; s += d * d; }
  red[tid] = s; __syncthreads();
  for (int o = 128; o > 0; o >>= 1) {
    if (tid < o) red[tid] += red[tid + o];
    __syncthreads();
  }
  if (tid == 0) stat = rsqrtf(red[0] * (1.0f / D_MODEL) + 1e-5f);
  __syncthreads();
  float rstd = stat;
#pragma unroll
  for (int i = 0; i < 4; ++i) {
    int cix = tid * 4 + i;
    x[(size_t)row * D_MODEL + cix] = (v[i] - mu) * rstd * g[cix] + bta[cix];
  }
}

// ---------------------------------------------------------------------------
// Host-side orchestration (all on `stream`, graph-capture safe).
// Workspace layout (floats): x | q | k | v | attn | tmp | hmid(2x)  ~128 MB
// ---------------------------------------------------------------------------
extern "C" void kernel_launch(void* const* d_in, const int* in_sizes, int n_in,
                              void* d_out, int out_size, void* d_ws,
                              size_t ws_size, hipStream_t stream) {
  (void)in_sizes; (void)n_in; (void)out_size; (void)ws_size;
  const float* tokens = (const float*)d_in[0];
  // d_in[1] = mask (causality is hardcoded)
  const float* qW = (const float*)d_in[2];
  const float* qB = (const float*)d_in[3];
  const float* kW = (const float*)d_in[4];
  const float* kB = (const float*)d_in[5];
  const float* vW = (const float*)d_in[6];
  const float* vB = (const float*)d_in[7];
  const float* oW = (const float*)d_in[8];
  const float* oB = (const float*)d_in[9];
  const float* ln1g = (const float*)d_in[10];
  const float* ln1b = (const float*)d_in[11];
  const float* w1 = (const float*)d_in[12];
  const float* b1 = (const float*)d_in[13];
  const float* w2 = (const float*)d_in[14];
  const float* b2 = (const float*)d_in[15];
  const float* ln2g = (const float*)d_in[16];
  const float* ln2b = (const float*)d_in[17];
  const float* lmW = (const float*)d_in[18];
  const float* lmB = (const float*)d_in[19];

  const size_t MT = (size_t)MROWS * D_MODEL;   // 4096*1024 elements
  float* x    = (float*)d_ws;
  float* q    = x + MT;
  float* k    = q + MT;
  float* v    = k + MT;
  float* attn = v + MT;
  float* tmp  = attn + MT;
  float* hmid = tmp + MT;                      // holds 2*MT (FF = 2D)

  posenc_kernel<<<MROWS, 256, 0, stream>>>(tokens, x);

  const dim3 gD(D_MODEL / BNG, MROWS / BMG);   // 8 x 32
  const dim3 gF(FFDIM / BNG, MROWS / BMG);     // 16 x 32
  const dim3 gV((NVOCAB + BNG - 1) / BNG, MROWS / BMG);
  const dim3 gA(SEQ / 64, NBATCH * NHEAD);

  for (int l = 0; l < NLAYER; ++l) {
    const size_t DD = (size_t)D_MODEL * D_MODEL;
    const size_t FD = (size_t)FFDIM * D_MODEL;
    gemm_f16w<<<gD, 256, 0, stream>>>(x, qW + l * DD, qB + l * D_MODEL, q,
                                      MROWS, D_MODEL, D_MODEL, 0);
    gemm_f16w<<<gD, 256, 0, stream>>>(x, kW + l * DD, kB + l * D_MODEL, k,
                                      MROWS, D_MODEL, D_MODEL, 0);
    gemm_f16w<<<gD, 256, 0, stream>>>(x, vW + l * DD, vB + l * D_MODEL, v,
                                      MROWS, D_MODEL, D_MODEL, 0);
    attention_fused<<<gA, 128, 0, stream>>>(q, k, v, attn);
    gemm_f16w<<<gD, 256, 0, stream>>>(attn, oW + l * DD, oB + l * D_MODEL, tmp,
                                      MROWS, D_MODEL, D_MODEL, 0);
    ln_residual<<<MROWS, 256, 0, stream>>>(x, tmp, ln1g + l * D_MODEL,
                                           ln1b + l * D_MODEL);
    gemm_f16w<<<gF, 256, 0, stream>>>(x, w1 + l * FD, b1 + l * FFDIM, hmid,
                                      MROWS, FFDIM, D_MODEL, 1);
    gemm_f16w<<<gD, 256, 0, stream>>>(hmid, w2 + l * FD, b2 + l * D_MODEL, tmp,
                                      MROWS, D_MODEL, FFDIM, 0);
    ln_residual<<<MROWS, 256, 0, stream>>>(x, tmp, ln2g + l * D_MODEL,
                                           ln2b + l * D_MODEL);
  }
  gemm_f16w<<<gV, 256, 0, stream>>>(x, lmW, lmB, (float*)d_out, MROWS, NVOCAB,
                                    D_MODEL, 0);
}
